// ANHPMultiHeadAttention_58703613002161
// MI455X (gfx1250) — compile-verified, matrix operand order checked by reading, not run on hardware
//
#include <hip/hip_runtime.h>
#include <hip/hip_bf16.h>

typedef __bf16 bf16;
typedef __attribute__((ext_vector_type(16))) __bf16 v16bf;
typedef __attribute__((ext_vector_type(8)))  float        v8f;
typedef __attribute__((ext_vector_type(4)))  unsigned int v4u;
typedef __attribute__((ext_vector_type(8)))  unsigned int v8u;

constexpr int NB   = 8;
constexpr int NS   = 1024;
constexpr int NF   = 1024;
constexpr int NH   = 1024;   // HID
constexpr int NHEAD= 8;
constexpr int NDH  = 128;    // head dim

// ---------------------------------------------------------------------------
// WMMA helpers (CDNA5 wave32, 16x16x32 bf16, f32 accumulate)
// ---------------------------------------------------------------------------
__device__ __forceinline__ v8f wmma_bf16(v16bf a, v16bf b, v8f c) {
  return __builtin_amdgcn_wmma_f32_16x16x32_bf16(false, a, false, b, (short)0, c,
                                                 false, false);
}

// Load a 16x32 A-fragment (or symmetric B-fragment) from a row-major tile
// with row stride `ldk` (elements).  Per ISA 7.12.2: lanes 0-15 hold row L,
// K = {half*8..half*8+7} U {16+half*8..16+half*8+7} with half = lane>>4.
// Both 16B chunks are contiguous -> two 128-bit loads.
__device__ __forceinline__ v16bf ld_frag(const bf16* base, int ldk) {
  const int lane = threadIdx.x & 31;
  const int r    = lane & 15;
  const int hh   = lane >> 4;
  const bf16* p  = base + (size_t)r * ldk + hh * 8;
  v4u lo = *reinterpret_cast<const v4u*>(p);
  v4u hi = *reinterpret_cast<const v4u*>(p + 16);
  v8u pk = __builtin_shufflevector(lo, hi, 0, 1, 2, 3, 4, 5, 6, 7);
  return __builtin_bit_cast(v16bf, pk);
}

// ---------------------------------------------------------------------------
// fp32 -> bf16 converts
// ---------------------------------------------------------------------------
__global__ void __launch_bounds__(256) cvt_x(const float* __restrict__ x,
                                             bf16* __restrict__ o, int n) {
  int i = blockIdx.x * 256 + threadIdx.x;
  if (i < n) o[i] = (bf16)x[i];
}

// Wt[mat][n][k] = W_mat[k][n]  (3 x 1024 x 1024)
__global__ void __launch_bounds__(256) cvt_wt(const float* __restrict__ Wq,
                                              const float* __restrict__ Wk,
                                              const float* __restrict__ Wv,
                                              bf16* __restrict__ wt) {
  int idx = blockIdx.x * 256 + threadIdx.x;
  if (idx >= 3 * NF * NH) return;
  int mat = idx >> 20;
  int rem = idx & 0xFFFFF;
  int n   = rem >> 10;
  int k   = rem & 1023;
  const float* W = (mat == 0) ? Wq : (mat == 1) ? Wk : Wv;
  wt[idx] = (bf16)W[(size_t)k * NH + n];
}

// ---------------------------------------------------------------------------
// QKV projection GEMM:  C[8192 x 3072] = X[8192 x 1024] @ Wt^T  (+ bias)
// 128x128 block tile, 8 waves (2x4), each wave 64x32 = 8 WMMA tiles.
// Q,K stored natural [b*S+s][HID]; V stored head-transposed Vt[(b,h)][dh][s].
// ---------------------------------------------------------------------------
__global__ void __launch_bounds__(256) qkv_gemm(
    const bf16* __restrict__ X,    // [8192][1024] bf16
    const bf16* __restrict__ Wt,   // [3][1024 n][1024 k] bf16
    const float* __restrict__ bq, const float* __restrict__ bk,
    const float* __restrict__ bv,
    bf16* __restrict__ Qb, bf16* __restrict__ Kb, bf16* __restrict__ Vt) {
  constexpr int TM = 128, TN = 128, TK = 32;
  __shared__ alignas(16) bf16 sA[TM * TK];
  __shared__ alignas(16) bf16 sB[TN * TK];

  const int mtile = blockIdx.x & 63;   // 8192/128 = 64
  const int ntile = blockIdx.x >> 6;   // 3072/128 = 24
  const int m0 = mtile * TM, n0 = ntile * TN;
  const int t  = threadIdx.x;
  const int w  = t >> 5;
  const int wr = w >> 2;               // 0..1  -> 64 rows
  const int wc = w & 3;                // 0..3  -> 32 cols
  const int lane = t & 31, half = lane >> 4, col = lane & 15;

  const v8f vz = {0.f, 0.f, 0.f, 0.f, 0.f, 0.f, 0.f, 0.f};
  v8f acc[4][2];
#pragma unroll
  for (int i = 0; i < 4; ++i)
#pragma unroll
    for (int j = 0; j < 2; ++j) acc[i][j] = vz;

  const int row = t >> 1, hf = t & 1;  // cooperative copy role (128 rows x 64B)

  for (int k0 = 0; k0 < NF; k0 += TK) {
    {  // A tile 128x32
      const v4u* src =
          reinterpret_cast<const v4u*>(X + (size_t)(m0 + row) * NF + k0 + hf * 16);
      v4u* dst = reinterpret_cast<v4u*>(sA + row * TK + hf * 16);
      dst[0] = src[0];
      dst[1] = src[1];
    }
    {  // B tile 128x32 (Wt already [n][k])
      const v4u* src =
          reinterpret_cast<const v4u*>(Wt + (size_t)(n0 + row) * NF + k0 + hf * 16);
      v4u* dst = reinterpret_cast<v4u*>(sB + row * TK + hf * 16);
      dst[0] = src[0];
      dst[1] = src[1];
    }
    if (k0 + TK < NF) {  // global_prefetch_b8 next K-step
      __builtin_prefetch(X + (size_t)(m0 + row) * NF + k0 + TK, 0, 1);
      __builtin_prefetch(Wt + (size_t)(n0 + row) * NF + k0 + TK, 0, 1);
    }
    __syncthreads();

    v16bf afr[4], bfr[2];
#pragma unroll
    for (int mi = 0; mi < 4; ++mi) afr[mi] = ld_frag(sA + (wr * 64 + mi * 16) * TK, TK);
#pragma unroll
    for (int ni = 0; ni < 2; ++ni) bfr[ni] = ld_frag(sB + (wc * 32 + ni * 16) * TK, TK);
#pragma unroll
    for (int mi = 0; mi < 4; ++mi)
#pragma unroll
      for (int ni = 0; ni < 2; ++ni)
        acc[mi][ni] = wmma_bf16(afr[mi], bfr[ni], acc[mi][ni]);
    __syncthreads();
  }

  // epilogue: bias add, bf16 store (C layout: elem i -> row i+8*half, col lane&15)
#pragma unroll
  for (int mi = 0; mi < 4; ++mi) {
#pragma unroll
    for (int ni = 0; ni < 2; ++ni) {
      int ncol = n0 + wc * 32 + ni * 16 + col;
      int mat  = ncol >> 10;
      int cc   = ncol & 1023;
      float bias = (mat == 0) ? bq[cc] : (mat == 1) ? bk[cc] : bv[cc];
#pragma unroll
      for (int i = 0; i < 8; ++i) {
        int mrow = m0 + wr * 64 + mi * 16 + i + half * 8;
        bf16 o = (bf16)(acc[mi][ni][i] + bias);
        if (mat == 0) {
          Qb[(size_t)mrow * NH + cc] = o;
        } else if (mat == 1) {
          Kb[(size_t)mrow * NH + cc] = o;
        } else {
          int hh2 = cc >> 7, dh = cc & 127;
          int bb = mrow >> 10, ss = mrow & 1023;
          Vt[(((size_t)bb * NHEAD + hh2) * NDH + dh) * NS + ss] = o;
        }
      }
    }
  }
}

// ---------------------------------------------------------------------------
// Flash-style attention with the reference's score transform:
//   y = exp(qk/sqrt(dh)) + (-128 if key>query else 0);  out = softmax(y) @ V
// Block = 4 waves (128 thr), 64 query rows, one (b,h). Stream keys 32 at a time.
// ---------------------------------------------------------------------------
__global__ void __launch_bounds__(128) attention(
    const bf16* __restrict__ Qb, const bf16* __restrict__ Kb,
    const bf16* __restrict__ Vt, float* __restrict__ Out) {
  __shared__ alignas(16) bf16 sK[32 * NDH];      // [key][dh]   8 KB
  __shared__ alignas(16) bf16 sV[NDH * 32];      // [dh][key]   8 KB
  __shared__ alignas(16) bf16 sP[4 * 16 * 32];   // per-wave P  4 KB

  const int qt = blockIdx.x & 15;   // 16 query tiles of 64
  const int bh = blockIdx.x >> 4;   // 0..63
  const int b  = bh >> 3, h = bh & 7;
  const int q0 = qt * 64;
  const int t  = threadIdx.x;
  const int w  = t >> 5, lane = t & 31;
  const int half = lane >> 4, col = lane & 15;

  // Q fragments for this wave's 16 rows (registers, reused over all keys)
  const bf16* qbase = Qb + ((size_t)(b * NS) + q0 + w * 16) * NH + h * NDH;
  v16bf qf[4];
#pragma unroll
  for (int d = 0; d < 4; ++d) qf[d] = ld_frag(qbase + d * 32, NH);

  const v8f vz = {0.f, 0.f, 0.f, 0.f, 0.f, 0.f, 0.f, 0.f};
  v8f acc[8];
#pragma unroll
  for (int d = 0; d < 8; ++d) acc[d] = vz;
  float mrun[8], lrun[8];
#pragma unroll
  for (int i = 0; i < 8; ++i) { mrun[i] = -3.0e38f; lrun[i] = 0.f; }

  const float scale = 0.08838834764831845f;  // 1/sqrt(128)
  const bf16* Kbh = Kb + (size_t)(b * NS) * NH + h * NDH;
  const bf16* Vbh = Vt + ((size_t)b * NHEAD + h) * NDH * NS;

  for (int kb = 0; kb < NS / 32; ++kb) {
    {  // K block [32 keys][128 dh] : thread -> (row=t>>2, 64B quarter)
      int krow = t >> 2, qq = t & 3;
      const v4u* src =
          reinterpret_cast<const v4u*>(Kbh + (size_t)(kb * 32 + krow) * NH + qq * 32);
      v4u* dst = reinterpret_cast<v4u*>(sK + krow * NDH + qq * 32);
      dst[0] = src[0]; dst[1] = src[1]; dst[2] = src[2]; dst[3] = src[3];
    }
    {  // V block [128 dh][32 keys] : thread t owns dh row t (64B)
      const v4u* src = reinterpret_cast<const v4u*>(Vbh + (size_t)t * NS + kb * 32);
      v4u* dst = reinterpret_cast<v4u*>(sV + t * 32);
      dst[0] = src[0]; dst[1] = src[1]; dst[2] = src[2]; dst[3] = src[3];
    }
    if (kb + 1 < NS / 32) {
      __builtin_prefetch(Kbh + (size_t)((kb + 1) * 32 + (t >> 2)) * NH, 0, 1);
      __builtin_prefetch(Vbh + (size_t)t * NS + (kb + 1) * 32, 0, 1);
    }
    __syncthreads();

    // S = Q @ K^T : two 16-key n-tiles, 4 K-steps over dh
    v8f sc[2] = {vz, vz};
#pragma unroll
    for (int nt = 0; nt < 2; ++nt)
#pragma unroll
      for (int d = 0; d < 4; ++d)
        sc[nt] = wmma_bf16(qf[d], ld_frag(sK + (nt * 16) * NDH + d * 32, NDH), sc[nt]);

    // score transform + online softmax (row r = i + 8*half, col = key)
    float p0[8], p1[8];
#pragma unroll
    for (int i = 0; i < 8; ++i) {
      int qrow = q0 + w * 16 + i + half * 8;
      int key0 = kb * 32 + col;
      int key1 = key0 + 16;
      float y0 = __expf(sc[0][i] * scale) + ((key0 > qrow) ? -128.0f : 0.0f);
      float y1 = __expf(sc[1][i] * scale) + ((key1 > qrow) ? -128.0f : 0.0f);
      float r = fmaxf(y0, y1);
      r = fmaxf(r, __shfl_xor(r, 1));
      r = fmaxf(r, __shfl_xor(r, 2));
      r = fmaxf(r, __shfl_xor(r, 4));
      r = fmaxf(r, __shfl_xor(r, 8));
      float mnew = fmaxf(mrun[i], r);
      float corr = __expf(mrun[i] - mnew);
      mrun[i] = mnew;
      lrun[i] *= corr;
#pragma unroll
      for (int d = 0; d < 8; ++d) acc[d][i] *= corr;
      float e0 = __expf(y0 - mnew);
      float e1 = __expf(y1 - mnew);
      p0[i] = e0; p1[i] = e1;
      float s2 = e0 + e1;
      s2 += __shfl_xor(s2, 1);
      s2 += __shfl_xor(s2, 2);
      s2 += __shfl_xor(s2, 4);
      s2 += __shfl_xor(s2, 8);
      lrun[i] += s2;
    }

    // C-layout -> A-fragment layout via wave-private LDS
    bf16* pw = sP + w * (16 * 32);
#pragma unroll
    for (int i = 0; i < 8; ++i) {
      int r = i + half * 8;
      pw[r * 32 + col]      = (bf16)p0[i];
      pw[r * 32 + 16 + col] = (bf16)p1[i];
    }
    v16bf pf = ld_frag(pw, 32);

    // acc += P @ V  (8 dh tiles, K = 32 keys)
#pragma unroll
    for (int d = 0; d < 8; ++d)
      acc[d] = wmma_bf16(pf, ld_frag(sV + (d * 16) * 32, 32), acc[d]);

    __syncthreads();
  }

  // normalize + store fp32 output [b][s][h*128 + dh]
#pragma unroll
  for (int i = 0; i < 8; ++i) {
    float inv = 1.0f / lrun[i];
    int qrow = q0 + w * 16 + i + half * 8;
    float* orow = Out + ((size_t)(b * NS) + qrow) * NH + h * NDH;
#pragma unroll
    for (int d = 0; d < 8; ++d) orow[d * 16 + col] = acc[d][i] * inv;
  }
}

// ---------------------------------------------------------------------------
// launch
// ---------------------------------------------------------------------------
extern "C" void kernel_launch(void* const* d_in, const int* in_sizes, int n_in,
                              void* d_out, int out_size, void* d_ws, size_t ws_size,
                              hipStream_t stream) {
  const float* queries = (const float*)d_in[0];
  const float* Wq = (const float*)d_in[1];
  const float* bq = (const float*)d_in[2];
  const float* Wk = (const float*)d_in[3];
  const float* bk = (const float*)d_in[4];
  const float* Wv = (const float*)d_in[5];
  const float* bv = (const float*)d_in[6];
  float* out = (float*)d_out;

  char* ws = (char*)d_ws;
  const size_t MiB = 1024 * 1024;
  bf16* xb = (bf16*)(ws);                 // 8192*1024 bf16 = 16 MiB
  bf16* wt = (bf16*)(ws + 16 * MiB);      // 3*1024*1024 bf16 = 6 MiB
  bf16* Qb = (bf16*)(ws + 22 * MiB);      // 16 MiB
  bf16* Kb = (bf16*)(ws + 38 * MiB);      // 16 MiB
  bf16* Vt = (bf16*)(ws + 54 * MiB);      // 16 MiB  (total 70 MiB)

  int nx = NB * NS * NF;
  cvt_x<<<(nx + 255) / 256, 256, 0, stream>>>(queries, xb, nx);
  cvt_wt<<<(3 * NF * NH + 255) / 256, 256, 0, stream>>>(Wq, Wk, Wv, wt);

  // 64 m-tiles x 24 n-tiles
  qkv_gemm<<<dim3(64 * 24), 256, 0, stream>>>(xb, wt, bq, bk, bv, Qb, Kb, Vt);

  // B*H*(S/64) = 1024 blocks, 128 threads
  attention<<<dim3(NB * NHEAD * (NS / 64)), 128, 0, stream>>>(Qb, Kb, Vt, out);
}
